// BasisFFN_71167608094942
// MI455X (gfx1250) — matrix operand order repaired and checked by reading, not compiled, not run on hardware
//
#include <hip/hip_runtime.h>
#include <hip/hip_bf16.h>

#define RESIDUAL_SCALE 0.1f
#define EPS_F 1e-8f

#define N_NEURONS 2048
#define D_MODEL   1024
#define D_FF      4096
#define N_BASIS   16
#define BASIS_RANK 64
#define BB 8
#define SS 2048
#define KK 8
#define TR_HIDDEN 256
#define M_TOT (BB*SS)   // 16384 rows

typedef __attribute__((ext_vector_type(16))) __bf16 v16bf;
typedef __attribute__((ext_vector_type(8)))  float  v8f;

// ---------- helpers ----------
__device__ __forceinline__ __bf16 f2bf(float f) {
  unsigned u = __float_as_uint(f);
  u += 0x7FFFu + ((u >> 16) & 1u);           // round-to-nearest-even
  unsigned short s = (unsigned short)(u >> 16);
  return __builtin_bit_cast(__bf16, s);
}

__device__ __forceinline__ v8f wmma_bf16(v16bf a, v16bf b, v8f c) {
  return __builtin_amdgcn_wmma_f32_16x16x32_bf16(
      false, a, false, b, (short)0, c, false, false);
}

// A fragment: 16x32 bf16 from row-major [M,K] source.
// Lane r holds row m0+r; VGPR i holds K pair (i<4?0:16)+half*8+2(i&3)
// => f[0..7] = row[k0+half*8 .. +7], f[8..15] = row[k0+16+half*8 .. +7]
// Two contiguous 16B chunks -> global_load_b128 pairs.
__device__ __forceinline__ v16bf load_frag_a(const __bf16* __restrict__ A,
                                             int lda, int m0, int k0) {
  int lane = threadIdx.x & 31;
  int half = lane >> 4, r = lane & 15;
  const __bf16* row = A + (size_t)(m0 + r) * lda + k0 + half * 8;
  v16bf f;
#pragma unroll
  for (int j = 0; j < 8; ++j) f[j] = row[j];
#pragma unroll
  for (int j = 0; j < 8; ++j) f[8 + j] = row[16 + j];
  return f;
}

// B fragment: 32x16 bf16 taken from TRANSPOSED storage Bt [N, K] row-major.
// Lane r holds column n0+r; f[0..7] = K k0+half*8..+7, f[8..15] = K k0+16+half*8..+7
// (same K order as load_frag_a) -> two contiguous 16B chunks (2x b128).
__device__ __forceinline__ v16bf load_frag_bT(const __bf16* __restrict__ Bt,
                                              int ldbt, int k0, int n0) {
  int lane = threadIdx.x & 31;
  int half = lane >> 4, r = lane & 15;
  const __bf16* p = Bt + (size_t)(n0 + r) * ldbt + k0 + half * 8;
  v16bf f;
#pragma unroll
  for (int j = 0; j < 8; ++j) f[j] = p[j];
#pragma unroll
  for (int j = 0; j < 8; ++j) f[8 + j] = p[16 + j];
  return f;
}

// C/D 16x16 f32: VGPR i -> row m0+half*8+i, col n0+(lane&15)
__device__ __forceinline__ void store_c_bf16(__bf16* __restrict__ C, int ldc,
                                             int m0, int n0, v8f acc) {
  int lane = threadIdx.x & 31;
  int half = lane >> 4, r = lane & 15;
#pragma unroll
  for (int i = 0; i < 8; ++i)
    C[(size_t)(m0 + half * 8 + i) * ldc + n0 + r] = f2bf(acc[i]);
}

// Transposed store: element (m0+half*8+i, n0+r) -> Ct[(n0+r)*ldct + m0+half*8+i]
// Per lane: 8 consecutive bf16 = one 16B store.
__device__ __forceinline__ void store_c_bf16_T(__bf16* __restrict__ Ct, int ldct,
                                               int m0, int n0, v8f acc) {
  int lane = threadIdx.x & 31;
  int half = lane >> 4, r = lane & 15;
  __bf16* p = Ct + (size_t)(n0 + r) * ldct + m0 + half * 8;
#pragma unroll
  for (int i = 0; i < 8; ++i) p[i] = f2bf(acc[i]);
}

// ---------- stage 1: sentence coefficients ----------
__global__ void coef_kernel(const int* __restrict__ nidx,
                            const float* __restrict__ w,
                            const float* __restrict__ coefA,
                            const float* __restrict__ coefB,
                            float* __restrict__ sentA,
                            float* __restrict__ sentB) {
  int b = blockIdx.x;
  const int N = SS * KK;
  __shared__ float sA[N_BASIS], sB[N_BASIS], sw;
  if (threadIdx.x < N_BASIS) { sA[threadIdx.x] = 0.f; sB[threadIdx.x] = 0.f; }
  if (threadIdx.x == 0) sw = 0.f;
  __syncthreads();
  float accA[N_BASIS], accB[N_BASIS], wsum = 0.f;
#pragma unroll
  for (int i = 0; i < N_BASIS; ++i) { accA[i] = 0.f; accB[i] = 0.f; }
  for (int n = threadIdx.x; n < N; n += blockDim.x) {
    float wn = w[(size_t)b * N + n];
    int id = nidx[(size_t)b * N + n];
    wsum += wn;
    const float* ca = coefA + (size_t)id * N_BASIS;
    const float* cb = coefB + (size_t)id * N_BASIS;
#pragma unroll
    for (int i = 0; i < N_BASIS; ++i) {
      accA[i] += wn * ca[i];
      accB[i] += wn * cb[i];
    }
  }
  atomicAdd(&sw, wsum);
#pragma unroll
  for (int i = 0; i < N_BASIS; ++i) {
    atomicAdd(&sA[i], accA[i]);
    atomicAdd(&sB[i], accB[i]);
  }
  __syncthreads();
  if (threadIdx.x < N_BASIS) {
    float inv = 1.0f / (sw + EPS_F);
    sentA[b * N_BASIS + threadIdx.x] = sA[threadIdx.x] * inv;
    sentB[b * N_BASIS + threadIdx.x] = sB[threadIdx.x] * inv;
  }
}

// ---------- stage 2: low-rank factor composition ----------
__global__ void build_A_kernel(const float* __restrict__ sentA,
                               const float* __restrict__ basisA,
                               __bf16* __restrict__ Abf) {
  size_t t = (size_t)blockIdx.x * blockDim.x + threadIdx.x; // B*1024*64
  int b = (int)(t >> 16);
  int dr = (int)(t & 65535);
  float acc = 0.f;
#pragma unroll
  for (int i = 0; i < N_BASIS; ++i)
    acc += sentA[b * N_BASIS + i] * basisA[(size_t)i * 65536 + dr];
  Abf[t] = f2bf(acc);                         // [B,1024,64] row-major (A operand)
}

// Produces Bmat^T: BbfT [B, 4096, 64]  (coalesced reads, scattered 2B writes ok: 4MB)
__global__ void build_B_kernel(const float* __restrict__ sentB,
                               const float* __restrict__ basisB,
                               __bf16* __restrict__ BbfT) {
  size_t t = (size_t)blockIdx.x * blockDim.x + threadIdx.x; // B*64*4096
  int b = (int)(t >> 18);
  int rf = (int)(t & 262143);
  int rr = rf >> 12, f = rf & 4095;
  float acc = 0.f;
#pragma unroll
  for (int i = 0; i < N_BASIS; ++i)
    acc += sentB[b * N_BASIS + i] * basisB[(size_t)i * 262144 + rf];
  BbfT[(size_t)b * 262144 + (size_t)f * BASIS_RANK + rr] = f2bf(acc);
}

// ---------- stage 3: W_up^T[b] (4096x1024) = (A[b] @ Bmat[b])^T ----------
__global__ void gemm_wup_kernel(const __bf16* __restrict__ Abf,
                                const __bf16* __restrict__ BbfT,
                                __bf16* __restrict__ WupT) {
  int b = blockIdx.z;
  int wave = threadIdx.x >> 5;
  int n0 = (blockIdx.x * 8 + wave) * 16;     // 0..4095
  int m0 = blockIdx.y * 16;                  // 0..1023
  const __bf16* A  = Abf  + (size_t)b * D_MODEL * BASIS_RANK;
  const __bf16* Bt = BbfT + (size_t)b * D_FF * BASIS_RANK;
  v8f acc = {};
#pragma unroll
  for (int k0 = 0; k0 < BASIS_RANK; k0 += 32)
    acc = wmma_bf16(load_frag_a(A, BASIS_RANK, m0, k0),
                    load_frag_bT(Bt, BASIS_RANK, k0, n0), acc);
  store_c_bf16_T(WupT + (size_t)b * D_FF * D_MODEL, D_MODEL, m0, n0, acc);
}

// ---------- stage 4: conversions ----------
__global__ void cvt_bf16_kernel(const float* __restrict__ in,
                                __bf16* __restrict__ out, size_t n) {
  size_t i = (size_t)blockIdx.x * blockDim.x + threadIdx.x;
  if (i < n) out[i] = f2bf(in[i]);
}

// f32 [R,C] row-major -> bf16 [C,R] row-major, LDS-tiled (coalesced both sides)
__global__ void cvt_bf16_T_kernel(const float* __restrict__ in,
                                  __bf16* __restrict__ out, int R, int C) {
  __shared__ float tile[32][33];
  int c0 = blockIdx.x * 32, r0 = blockIdx.y * 32;
  int tx = threadIdx.x & 31, ty = threadIdx.x >> 5;   // 256 thr = 8x32
#pragma unroll
  for (int j = 0; j < 32; j += 8)
    tile[ty + j][tx] = in[(size_t)(r0 + ty + j) * C + c0 + tx];
  __syncthreads();
#pragma unroll
  for (int j = 0; j < 32; j += 8)
    out[(size_t)(c0 + ty + j) * R + r0 + tx] = f2bf(tile[tx][ty + j]);
}

__global__ void token_sig_kernel(const float* __restrict__ sel,
                                 const float* __restrict__ w,
                                 __bf16* __restrict__ ts) {
  int bs = blockIdx.x;                       // [0, B*S)
  float wk[KK];
#pragma unroll
  for (int k = 0; k < KK; ++k) wk[k] = w[(size_t)bs * KK + k];
  const float* base = sel + (size_t)bs * KK * D_MODEL;
  for (int d = threadIdx.x; d < D_MODEL; d += blockDim.x) {
    float acc = 0.f;
#pragma unroll
    for (int k = 0; k < KK; ++k)
      acc += wk[k] * base[(size_t)k * D_MODEL + d];
    ts[(size_t)bs * D_MODEL + d] = f2bf(acc);
  }
}

// ---------- stage 5: t1 = relu(ts @ tr_w1 + b1), w1 transposed [256,1024] ----------
__global__ void gemm_t1_kernel(const __bf16* __restrict__ ts,
                               const __bf16* __restrict__ w1T,
                               const float* __restrict__ b1,
                               __bf16* __restrict__ t1) {
  int wave = threadIdx.x >> 5;
  int n0 = wave * 32;                        // 8 waves * 2 tiles * 16 = 256 = N
  int m0 = blockIdx.y * 16;
  v8f acc0 = {}, acc1 = {};
#pragma unroll 4
  for (int k0 = 0; k0 < D_MODEL; k0 += 32) {
    v16bf a = load_frag_a(ts, D_MODEL, m0, k0);
    acc0 = wmma_bf16(a, load_frag_bT(w1T, D_MODEL, k0, n0), acc0);
    acc1 = wmma_bf16(a, load_frag_bT(w1T, D_MODEL, k0, n0 + 16), acc1);
  }
  int lane = threadIdx.x & 31;
  int half = lane >> 4, r = lane & 15;
#pragma unroll
  for (int tile = 0; tile < 2; ++tile) {
    v8f acc = tile ? acc1 : acc0;
    int col = n0 + tile * 16 + r;
    float bias = b1[col];
#pragma unroll
    for (int i = 0; i < 8; ++i) {
      float v = acc[i] + bias;
      t1[(size_t)(m0 + half * 8 + i) * TR_HIDDEN + col] = f2bf(v > 0.f ? v : 0.f);
    }
  }
}

// ---------- stage 6: h = gelu(x @ W_up) + 0.1*(t1 @ tr_w2 + b2) ----------
__global__ void gemm_h_kernel(const __bf16* __restrict__ xbf,
                              const __bf16* __restrict__ WupT,
                              const __bf16* __restrict__ t1,
                              const __bf16* __restrict__ w2T,
                              const float* __restrict__ b2,
                              __bf16* __restrict__ h) {
  int wave = threadIdx.x >> 5;
  int n0 = (blockIdx.x * 16 + wave * 2) * 16;  // N = 4096, 256 cols per block
  int m0 = blockIdx.y * 16;
  int b = m0 >> 11;
  const __bf16* Wt = WupT + (size_t)b * D_FF * D_MODEL;
  v8f c00 = {}, c01 = {};
#pragma unroll 4
  for (int k0 = 0; k0 < D_MODEL; k0 += 32) {
    if (k0 + 128 < D_MODEL) {
      __builtin_prefetch(xbf + (size_t)m0 * D_MODEL + k0 + 128, 0, 0);
      __builtin_prefetch(Wt + (size_t)n0 * D_MODEL + k0 + 128, 0, 0);
    }
    v16bf a = load_frag_a(xbf, D_MODEL, m0, k0);
    c00 = wmma_bf16(a, load_frag_bT(Wt, D_MODEL, k0, n0), c00);
    c01 = wmma_bf16(a, load_frag_bT(Wt, D_MODEL, k0, n0 + 16), c01);
  }
  v8f r0 = {}, r1 = {};
#pragma unroll
  for (int k0 = 0; k0 < TR_HIDDEN; k0 += 32) {
    v16bf a = load_frag_a(t1, TR_HIDDEN, m0, k0);
    r0 = wmma_bf16(a, load_frag_bT(w2T, TR_HIDDEN, k0, n0), r0);
    r1 = wmma_bf16(a, load_frag_bT(w2T, TR_HIDDEN, k0, n0 + 16), r1);
  }
  int lane = threadIdx.x & 31;
  int half = lane >> 4, r = lane & 15;
#pragma unroll
  for (int tile = 0; tile < 2; ++tile) {
    v8f cg = tile ? c01 : c00;
    v8f cr = tile ? r1 : r0;
    int col = n0 + tile * 16 + r;
    float bias = b2[col];
#pragma unroll
    for (int i = 0; i < 8; ++i) {
      float g = cg[i];
      g = 0.5f * g * (1.0f + erff(g * 0.70710678118654752f));  // exact GELU
      float v = g + RESIDUAL_SCALE * (cr[i] + bias);
      h[(size_t)(m0 + half * 8 + i) * D_FF + col] = f2bf(v);
    }
  }
}

// ---------- stage 7: out = h @ down_w + down_b, down_w transposed [1024,4096] ----------
__global__ void gemm_out_kernel(const __bf16* __restrict__ h,
                                const __bf16* __restrict__ dwT,
                                const float* __restrict__ db,
                                float* __restrict__ out) {
  int wave = threadIdx.x >> 5;
  int n0 = (blockIdx.x * 16 + wave * 2) * 16;  // N = 1024
  int m0 = blockIdx.y * 16;
  v8f c0 = {}, c1 = {};
#pragma unroll 4
  for (int k0 = 0; k0 < D_FF; k0 += 32) {
    if (k0 + 128 < D_FF) {
      __builtin_prefetch(h + (size_t)m0 * D_FF + k0 + 128, 0, 0);
      __builtin_prefetch(dwT + (size_t)n0 * D_FF + k0 + 128, 0, 0);
    }
    v16bf a = load_frag_a(h, D_FF, m0, k0);
    c0 = wmma_bf16(a, load_frag_bT(dwT, D_FF, k0, n0), c0);
    c1 = wmma_bf16(a, load_frag_bT(dwT, D_FF, k0, n0 + 16), c1);
  }
  int lane = threadIdx.x & 31;
  int half = lane >> 4, r = lane & 15;
#pragma unroll
  for (int tile = 0; tile < 2; ++tile) {
    v8f acc = tile ? c1 : c0;
    int col = n0 + tile * 16 + r;
    float bias = db[col];
#pragma unroll
    for (int i = 0; i < 8; ++i)
      out[(size_t)(m0 + half * 8 + i) * D_MODEL + col] = acc[i] + bias;
  }
}

// ---------- host launch ----------
extern "C" void kernel_launch(void* const* d_in, const int* in_sizes, int n_in,
                              void* d_out, int out_size, void* d_ws, size_t ws_size,
                              hipStream_t stream) {
  (void)in_sizes; (void)n_in; (void)out_size; (void)ws_size;
  const float* x      = (const float*)d_in[0];
  const float* sel    = (const float*)d_in[1];
  const int*   nidx   = (const int*)  d_in[2];
  const float* nw     = (const float*)d_in[3];
  const float* basisA = (const float*)d_in[4];
  const float* basisB = (const float*)d_in[5];
  const float* coefA  = (const float*)d_in[6];
  const float* coefB  = (const float*)d_in[7];
  const float* tr_w1  = (const float*)d_in[8];
  const float* tr_b1  = (const float*)d_in[9];
  const float* tr_w2  = (const float*)d_in[10];
  const float* tr_b2  = (const float*)d_in[11];
  const float* down_w = (const float*)d_in[12];
  const float* down_b = (const float*)d_in[13];
  float* out = (float*)d_out;

  char* ws = (char*)d_ws;
  size_t off = 0;
  auto alloc = [&](size_t bytes) -> char* {
    char* p = ws + off;
    off = (off + bytes + 255) & ~(size_t)255;
    return p;
  };
  float*  sentA = (float*) alloc(BB * N_BASIS * sizeof(float));
  float*  sentB = (float*) alloc(BB * N_BASIS * sizeof(float));
  __bf16* Abf   = (__bf16*)alloc((size_t)BB * D_MODEL * BASIS_RANK * 2);
  __bf16* BbfT  = (__bf16*)alloc((size_t)BB * D_FF * BASIS_RANK * 2);
  __bf16* WupT  = (__bf16*)alloc((size_t)BB * D_FF * D_MODEL * 2);
  __bf16* xbf   = (__bf16*)alloc((size_t)M_TOT * D_MODEL * 2);
  __bf16* tsbf  = (__bf16*)alloc((size_t)M_TOT * D_MODEL * 2);
  __bf16* t1bf  = (__bf16*)alloc((size_t)M_TOT * TR_HIDDEN * 2);
  __bf16* hbf   = (__bf16*)alloc((size_t)M_TOT * D_FF * 2);
  __bf16* w1T   = (__bf16*)alloc((size_t)TR_HIDDEN * D_MODEL * 2);
  __bf16* w2T   = (__bf16*)alloc((size_t)D_FF * TR_HIDDEN * 2);
  __bf16* dwT   = (__bf16*)alloc((size_t)D_MODEL * D_FF * 2);

  // 1. sentence coefficients
  coef_kernel<<<BB, 256, 0, stream>>>(nidx, nw, coefA, coefB, sentA, sentB);

  // 2. compose low-rank factors (A row-major, Bmat transposed)
  build_A_kernel<<<(BB * D_MODEL * BASIS_RANK) / 256, 256, 0, stream>>>(sentA, basisA, Abf);
  build_B_kernel<<<(BB * BASIS_RANK * D_FF) / 256, 256, 0, stream>>>(sentB, basisB, BbfT);

  // 3. W_up^T[b] = (A[b] @ Bmat[b])^T  (WMMA)
  gemm_wup_kernel<<<dim3(D_FF / 128, D_MODEL / 16, BB), 256, 0, stream>>>(Abf, BbfT, WupT);

  // 4. bf16 staging (weights transposed) + token_sig
  cvt_bf16_kernel<<<(unsigned)(((size_t)M_TOT * D_MODEL) / 256), 256, 0, stream>>>(
      x, xbf, (size_t)M_TOT * D_MODEL);
  cvt_bf16_T_kernel<<<dim3(TR_HIDDEN / 32, D_MODEL / 32), 256, 0, stream>>>(
      tr_w1, w1T, D_MODEL, TR_HIDDEN);
  cvt_bf16_T_kernel<<<dim3(D_FF / 32, TR_HIDDEN / 32), 256, 0, stream>>>(
      tr_w2, w2T, TR_HIDDEN, D_FF);
  cvt_bf16_T_kernel<<<dim3(D_MODEL / 32, D_FF / 32), 256, 0, stream>>>(
      down_w, dwT, D_FF, D_MODEL);
  token_sig_kernel<<<M_TOT, 256, 0, stream>>>(sel, nw, tsbf);

  // 5. t1 = relu(token_sig @ tr_w1 + b1)
  gemm_t1_kernel<<<dim3(1, M_TOT / 16), 256, 0, stream>>>(tsbf, w1T, tr_b1, t1bf);

  // 6. h = gelu(x @ W_up) + 0.1*(t1 @ tr_w2 + b2)
  gemm_h_kernel<<<dim3(D_FF / 256, M_TOT / 16), 256, 0, stream>>>(
      xbf, WupT, t1bf, w2T, tr_b2, hbf);

  // 7. out = h @ down_w + down_b
  gemm_out_kernel<<<dim3(D_MODEL / 256, M_TOT / 16), 256, 0, stream>>>(
      hbf, dwT, down_b, out);
}
